// MSTB_30313879175424
// MI455X (gfx1250) — matrix operand order repaired and checked by reading, not compiled, not run on hardware
//
#include <hip/hip_runtime.h>

// ---------------- problem constants ----------------
#define HH    192
#define WW    192
#define DIMC  128
#define C2    64
#define H2    4
#define WSZ   16
#define OWS   24
#define PADB  4
#define NW    12          // windows per side
#define NWIN  144         // 12*12
#define NTOK  (HH*WW)     // 36864
#define NQ1   256         // WSZ*WSZ
#define NK1   576         // OWS*OWS
#define NPROW (NWIN*NK1)  // 82944
#define HID   256
#define CHG   128         // HID/2
#define CANV  200         // HH + 2*PADB
#define SCL   0.25f       // (C2/H2)^-0.5 = 16^-0.5

typedef __attribute__((ext_vector_type(16))) _Float16 v16h;
typedef __attribute__((ext_vector_type(8)))  _Float16 v8h;
typedef __attribute__((ext_vector_type(8)))  float    v8f;

union frag16 { v16h v; uint4 q[2]; _Float16 h[16]; };
union h8u    { v8h  v; uint4 q;    _Float16 h[8];  };

__device__ __forceinline__ int imin(int a, int b) { return a < b ? a : b; }
__device__ __forceinline__ int imax(int a, int b) { return a > b ? a : b; }

// ---------------- weight convert f32 -> f16 transposed (out[n][k] = in[k][n]) ----------------
__global__ void wcvtT_k(const float* __restrict__ in, _Float16* __restrict__ outT, int K, int N) {
  int idx = blockIdx.x * 256 + threadIdx.x;
  if (idx >= K * N) return;
  int n = idx % N, k = idx / N;
  outT[(size_t)n * K + k] = (_Float16)in[idx];
}

// ---------------- fused LN1 + window partition/split ----------------
__global__ __launch_bounds__(128) void ln1_part_k(
    const float* __restrict__ x, const float* __restrict__ w, const float* __restrict__ b,
    _Float16* __restrict__ xa, _Float16* __restrict__ xbimg) {
  int tok  = blockIdx.x * 4 + (threadIdx.x >> 5);
  int lane = threadIdx.x & 31;
  const float* xr = x + (size_t)tok * DIMC;
  float v[4]; float s = 0.f;
  #pragma unroll
  for (int j = 0; j < 4; ++j) { v[j] = xr[lane + j * 32]; s += v[j]; }
  #pragma unroll
  for (int off = 16; off; off >>= 1) s += __shfl_xor(s, off, 32);
  float mu = s * (1.f / 128.f);
  float q = 0.f;
  #pragma unroll
  for (int j = 0; j < 4; ++j) { float d = v[j] - mu; q += d * d; }
  #pragma unroll
  for (int off = 16; off; off >>= 1) q += __shfl_xor(q, off, 32);
  float rinv = rsqrtf(q * (1.f / 128.f) + 1e-5f);
  int y = tok / WW, xc = tok % WW;
  int win = (y / WSZ) * NW + (xc / WSZ);
  int loc = (y % WSZ) * WSZ + (xc % WSZ);
  #pragma unroll
  for (int j = 0; j < 4; ++j) {
    int c = lane + j * 32;
    float o = (v[j] - mu) * rinv * w[c] + b[c];
    if (c < C2) xa[((size_t)win * NQ1 + loc) * C2 + c] = (_Float16)o;
    else        xbimg[(size_t)tok * C2 + (c - C2)] = (_Float16)o;
  }
}

// ---------------- plain LN -> f16 (token-major) ----------------
__global__ __launch_bounds__(128) void ln_f16_k(
    const float* __restrict__ x, const float* __restrict__ w, const float* __restrict__ b,
    _Float16* __restrict__ out) {
  int tok  = blockIdx.x * 4 + (threadIdx.x >> 5);
  int lane = threadIdx.x & 31;
  const float* xr = x + (size_t)tok * DIMC;
  float v[4]; float s = 0.f;
  #pragma unroll
  for (int j = 0; j < 4; ++j) { v[j] = xr[lane + j * 32]; s += v[j]; }
  #pragma unroll
  for (int off = 16; off; off >>= 1) s += __shfl_xor(s, off, 32);
  float mu = s * (1.f / 128.f);
  float q = 0.f;
  #pragma unroll
  for (int j = 0; j < 4; ++j) { float d = v[j] - mu; q += d * d; }
  #pragma unroll
  for (int off = 16; off; off >>= 1) q += __shfl_xor(q, off, 32);
  float rinv = rsqrtf(q * (1.f / 128.f) + 1e-5f);
  #pragma unroll
  for (int j = 0; j < 4; ++j) {
    int c = lane + j * 32;
    out[(size_t)tok * DIMC + c] = (_Float16)((v[j] - mu) * rinv * w[c] + b[c]);
  }
}

// ---------------- unfold overlapping patches (8 channels / thread, b128) ----------------
__global__ void unfold_k(const _Float16* __restrict__ xbimg, _Float16* __restrict__ xbp) {
  size_t idx = (size_t)blockIdx.x * 256 + threadIdx.x;   // NPROW * 8
  int c8 = (int)(idx & 7);
  size_t row = idx >> 3;
  int p = (int)(row % NK1), win = (int)(row / NK1);
  int ty = p / OWS, tx = p % OWS;
  int wr = win / NW, wc = win % NW;
  int gy = wr * WSZ + ty - PADB, gx = wc * WSZ + tx - PADB;
  uint4 v; v.x = v.y = v.z = v.w = 0u;
  if (gy >= 0 && gy < HH && gx >= 0 && gx < WW)
    v = *(const uint4*)&xbimg[((size_t)gy * WW + gx) * C2 + c8 * 8];
  *(uint4*)&xbp[row * C2 + c8 * 8] = v;
}

// ---------------- generic WMMA GEMM: C = A(f16,MxK) * BT(f16,NxK) [+bias][+resid] ----------------
// 256 threads (8 waves); block tile 128x64; K step 32. All LDS traffic is b128.
__global__ __launch_bounds__(256) void gemm_wmma_k(
    const _Float16* __restrict__ A, const _Float16* __restrict__ BT,
    const float* __restrict__ bias, const float* __restrict__ resid,
    float* __restrict__ outF, _Float16* __restrict__ outH,
    int M, int N, int K) {
  __shared__ _Float16 As[128 * 40];   // row stride 40 halfs (80B) to spread banks
  __shared__ _Float16 Bs[64 * 40];
  const int tid = threadIdx.x;
  const int lane = tid & 31, wv = tid >> 5;
  const int m0 = blockIdx.y * 128;
  const int n0 = blockIdx.x * 64;
  const int mrow = lane & 15, half = lane >> 4;
  v8f acc[4] = {};
  for (int k0 = 0; k0 < K; k0 += 32) {
    #pragma unroll
    for (int it = 0; it < 2; ++it) {          // A: 128x32 halfs = 512 uint4
      int idx = tid + it * 256;
      int r = idx >> 2, qo = idx & 3;
      *(uint4*)&As[r * 40 + qo * 8] =
          *(const uint4*)&A[(size_t)(m0 + r) * K + k0 + qo * 8];
    }
    {                                         // B: 64x32 halfs = 256 uint4
      int r = tid >> 2, qo = tid & 3;
      *(uint4*)&Bs[r * 40 + qo * 8] =
          *(const uint4*)&BT[(size_t)(n0 + r) * K + k0 + qo * 8];
    }
    __syncthreads();
    if (k0 + 32 < K) {   // global_prefetch_b8 of next K tile
      __builtin_prefetch(&A[(size_t)(m0 + (tid >> 1)) * K + k0 + 32 + (tid & 1) * 16], 0, 1);
      __builtin_prefetch(&BT[(size_t)(n0 + (tid & 63)) * K + k0 + 32 + ((tid >> 6) & 1) * 16], 0, 1);
    }
    frag16 af;
    af.q[0] = *(const uint4*)&As[(wv * 16 + mrow) * 40 + half * 8];
    af.q[1] = *(const uint4*)&As[(wv * 16 + mrow) * 40 + 16 + half * 8];
    #pragma unroll
    for (int t = 0; t < 4; ++t) {
      frag16 bf;
      int n = t * 16 + mrow;
      bf.q[0] = *(const uint4*)&Bs[n * 40 + half * 8];
      bf.q[1] = *(const uint4*)&Bs[n * 40 + 16 + half * 8];
      acc[t] = __builtin_amdgcn_wmma_f32_16x16x32_f16(
          false, af.v, false, bf.v, (short)0, acc[t], false, false);
    }
    __syncthreads();
  }
  #pragma unroll
  for (int t = 0; t < 4; ++t) {
    int n = n0 + t * 16 + mrow;
    float bv = bias ? bias[n] : 0.f;
    #pragma unroll
    for (int r = 0; r < 8; ++r) {
      int m = m0 + wv * 16 + r + half * 8;
      size_t o = (size_t)m * N + n;
      float v = acc[t][r] + bv;
      if (resid) v += resid[o];
      if (outF) outF[o] = v;
      if (outH) outH[o] = (_Float16)v;
    }
  }
}

// ---------------- cross attention (1 wave per (window, head, 16-query tile)) ----------------
// head dim 16 zero-padded to K=32. kv layout: [row][128], k=0..63, v=64..127.
// LDS: scores/probs f16 [16][NK1], V transposed f16 [dim16][NK1].
__global__ __launch_bounds__(32) void attn_wmma_k(
    const _Float16* __restrict__ qb, const _Float16* __restrict__ kvb,
    const _Float16* __restrict__ residb,
    const int* __restrict__ rpi, const float* __restrict__ rpb,
    _Float16* __restrict__ outH, float* __restrict__ outF,
    int NQ, int NK, int biasMode) {
  __shared__ _Float16 sch[16][NK1];   // 18KB
  __shared__ _Float16 vth[16][NK1];   // 18KB
  const int lane = threadIdx.x;
  const int qt = blockIdx.x, h = blockIdx.y, win = blockIdx.z;
  const int mrow = lane & 15, half = lane >> 4;

  // Q fragment: real dims are contiguous low 16 of padded K
  frag16 af;
  {
    size_t qrow = (size_t)win * NQ + qt * 16 + mrow;
    af.q[0] = *(const uint4*)&qb[qrow * C2 + h * 16 + half * 8];  // K=half*8..+7 real
    af.q[1].x = af.q[1].y = af.q[1].z = af.q[1].w = 0u;           // K=16..31 pad
    // zero the pad half of q[0]'s upper group? layout: elements 0..7 real, 8..15 pad -> q[1]=0 OK
  }
  for (int nt = 0; nt < NK / 16; ++nt) {
    // stage V tile (16 keys x 16 dims) transposed into LDS: 1 b128 load + 8 b16 stores/lane
    {
      int key = nt * 16 + (lane >> 1);
      int d0 = (lane & 1) * 8;
      h8u vv;
      vv.q = *(const uint4*)&kvb[((size_t)win * NK + key) * 128 + 64 + h * 16 + d0];
      #pragma unroll
      for (int j = 0; j < 8; ++j) vth[d0 + j][key] = vv.h[j];
    }
    // K fragment: lane half h carries K(dim)=8h..8h+7 in elements 0..7; elements 8..15 are the zero pad
    frag16 bf;
    int key = nt * 16 + mrow;
    bf.q[0] = *(const uint4*)&kvb[((size_t)win * NK + key) * 128 + h * 16 + half * 8];
    bf.q[1].x = bf.q[1].y = bf.q[1].z = bf.q[1].w = 0u;
    v8f c = {};
    c = __builtin_amdgcn_wmma_f32_16x16x32_f16(false, af.v, false, bf.v, (short)0, c, false, false);
    #pragma unroll
    for (int r = 0; r < 8; ++r) {
      int m = r + half * 8;
      int q = qt * 16 + m;
      int k = nt * 16 + mrow;
      int bi = (biasMode == 1) ? rpi[q * NK1 + k] : rpi[k * NK1 + q];
      sch[m][k] = (_Float16)(c[r] * SCL + rpb[bi * H2 + h]);
    }
  }
  __syncthreads();
  // row softmax in f16 LDS (lanes 0..15; v8h vector ops)
  if (lane < 16) {
    float mx = -1e30f;
    for (int k = 0; k < NK; k += 8) {
      h8u xv; xv.v = *(const v8h*)&sch[lane][k];
      #pragma unroll
      for (int j = 0; j < 8; ++j) mx = fmaxf(mx, (float)xv.h[j]);
    }
    float sum = 0.f;
    for (int k = 0; k < NK; k += 8) {
      h8u xv; xv.v = *(const v8h*)&sch[lane][k];
      #pragma unroll
      for (int j = 0; j < 8; ++j) { float e = __expf((float)xv.h[j] - mx); sum += e; xv.h[j] = (_Float16)e; }
      *(v8h*)&sch[lane][k] = xv.v;
    }
    float inv = 1.f / sum;
    for (int k = 0; k < NK; k += 8) {
      h8u xv; xv.v = *(const v8h*)&sch[lane][k];
      #pragma unroll
      for (int j = 0; j < 8; ++j) xv.h[j] = (_Float16)((float)xv.h[j] * inv);
      *(v8h*)&sch[lane][k] = xv.v;
    }
  }
  __syncthreads();
  // O = P V : pure b128 LDS traffic into WMMA
  v8f acc = {};
  for (int kc = 0; kc < NK; kc += 32) {
    frag16 pf, vf;
    pf.q[0] = *(const uint4*)&sch[mrow][kc + half * 8];
    pf.q[1] = *(const uint4*)&sch[mrow][kc + 16 + half * 8];
    vf.q[0] = *(const uint4*)&vth[mrow][kc + half * 8];
    vf.q[1] = *(const uint4*)&vth[mrow][kc + 16 + half * 8];
    acc = __builtin_amdgcn_wmma_f32_16x16x32_f16(false, pf.v, false, vf.v, (short)0, acc, false, false);
  }
  #pragma unroll
  for (int r = 0; r < 8; ++r) {
    int q = qt * 16 + r + half * 8;
    size_t o = ((size_t)win * NQ + q) * C2 + h * 16 + mrow;
    float v = acc[r] + (float)residb[o];
    if (outH) outH[o] = (_Float16)v;
    else      outF[o] = v;
  }
}

// ---------------- deterministic fold (gather, <=4 windows; 8 channels/thread) ----------------
__global__ void fold_k(const float* __restrict__ bb, float* __restrict__ canvas) {
  size_t idx = (size_t)blockIdx.x * 256 + threadIdx.x;  // CANV*CANV*8
  int c8 = (int)(idx & 7);
  int pix = (int)(idx >> 3);
  int x = pix % CANV, y = pix / CANV;
  int wr_lo = (y >= OWS) ? ((y - (OWS - 1) + (WSZ - 1)) / WSZ) : 0;
  int wr_hi = imin(NW - 1, y / WSZ);
  int wc_lo = (x >= OWS) ? ((x - (OWS - 1) + (WSZ - 1)) / WSZ) : 0;
  int wc_hi = imin(NW - 1, x / WSZ);
  float4 s0 = make_float4(0.f, 0.f, 0.f, 0.f), s1 = s0;
  for (int wr = wr_lo; wr <= wr_hi; ++wr) {
    int ty = y - wr * WSZ;
    if (ty < 0 || ty >= OWS) continue;
    for (int wc = wc_lo; wc <= wc_hi; ++wc) {
      int tx = x - wc * WSZ;
      if (tx < 0 || tx >= OWS) continue;
      const float4* p = (const float4*)&bb[(((size_t)(wr * NW + wc)) * NK1 + ty * OWS + tx) * C2 + c8 * 8];
      float4 a0 = p[0], a1 = p[1];
      s0.x += a0.x; s0.y += a0.y; s0.z += a0.z; s0.w += a0.w;
      s1.x += a1.x; s1.y += a1.y; s1.z += a1.z; s1.w += a1.w;
    }
  }
  float4* o = (float4*)&canvas[((size_t)pix) * C2 + c8 * 8];
  o[0] = s0; o[1] = s1;
}

// ---------------- bicubic resize + concat -> x_cat f16 (8 channels/thread) ----------------
__device__ __forceinline__ float cubicw(float x) {
  float ax = fabsf(x);
  const float A = -0.75f;
  if (ax <= 1.f) return ((A + 2.f) * ax - (A + 3.f)) * ax * ax + 1.f;
  if (ax < 2.f)  return A * (ax * (ax * (ax - 5.f) + 8.f) - 4.f);
  return 0.f;
}
__global__ void concat_k(const _Float16* __restrict__ aw, const float* __restrict__ canvas,
                         _Float16* __restrict__ xcat) {
  size_t idx = (size_t)blockIdx.x * 256 + threadIdx.x;  // NTOK * 16
  int g = (int)(idx & 15);          // group of 8 channels
  int tok = (int)(idx >> 4);
  int y = tok / WW, x = tok % WW;
  if (g < 8) {
    int win = (y / WSZ) * NW + (x / WSZ);
    int loc = (y % WSZ) * WSZ + (x % WSZ);
    *(uint4*)&xcat[(size_t)tok * DIMC + g * 8] =
        *(const uint4*)&aw[((size_t)win * NQ1 + loc) * C2 + g * 8];
  } else {
    int c0 = (g - 8) * 8;
    float sy = (y + 0.5f) * ((float)CANV / (float)HH) - 0.5f;
    float sx = (x + 0.5f) * ((float)CANV / (float)WW) - 0.5f;
    float fy = floorf(sy), fx = floorf(sx);
    float s[8] = {0.f, 0.f, 0.f, 0.f, 0.f, 0.f, 0.f, 0.f};
    #pragma unroll
    for (int a = 0; a < 4; ++a) {
      float py = fy + (float)(a - 1);
      float wy = cubicw(sy - py);
      int iy = imin(imax((int)py, 0), CANV - 1);
      #pragma unroll
      for (int bq = 0; bq < 4; ++bq) {
        float px = fx + (float)(bq - 1);
        float wx = cubicw(sx - px);
        int ix = imin(imax((int)px, 0), CANV - 1);
        float wgt = wy * wx;
        const float4* p = (const float4*)&canvas[((size_t)iy * CANV + ix) * C2 + c0];
        float4 a0 = p[0], a1 = p[1];
        s[0] += wgt * a0.x; s[1] += wgt * a0.y; s[2] += wgt * a0.z; s[3] += wgt * a0.w;
        s[4] += wgt * a1.x; s[5] += wgt * a1.y; s[6] += wgt * a1.z; s[7] += wgt * a1.w;
      }
    }
    h8u o;
    #pragma unroll
    for (int j = 0; j < 8; ++j) o.h[j] = (_Float16)s[j];
    *(uint4*)&xcat[(size_t)tok * DIMC + C2 + c0] = o.q;
  }
}

// ---------------- GSFN: LN of gate half -> conv input (NCHW) ----------------
__global__ __launch_bounds__(128) void sgnorm_k(
    const float* __restrict__ u, const float* __restrict__ w, const float* __restrict__ b,
    float* __restrict__ convin) {
  int tok  = blockIdx.x * 4 + (threadIdx.x >> 5);
  int lane = threadIdx.x & 31;
  const float* xr = u + (size_t)tok * HID + CHG;
  float v[4]; float s = 0.f;
  #pragma unroll
  for (int j = 0; j < 4; ++j) { v[j] = xr[lane + j * 32]; s += v[j]; }
  #pragma unroll
  for (int off = 16; off; off >>= 1) s += __shfl_xor(s, off, 32);
  float mu = s * (1.f / 128.f);
  float q = 0.f;
  #pragma unroll
  for (int j = 0; j < 4; ++j) { float d = v[j] - mu; q += d * d; }
  #pragma unroll
  for (int off = 16; off; off >>= 1) q += __shfl_xor(q, off, 32);
  float rinv = rsqrtf(q * (1.f / 128.f) + 1e-5f);
  #pragma unroll
  for (int j = 0; j < 4; ++j) {
    int c = lane + j * 32;
    convin[(size_t)c * NTOK + tok] = (v[j] - mu) * rinv * w[c] + b[c];
  }
}

// ---------------- depthwise 3x3 conv + gate multiply -> g f16 ----------------
__global__ void convgate_k(const float* __restrict__ convin, const float* __restrict__ cw,
                           const float* __restrict__ cb, const float* __restrict__ u,
                           _Float16* __restrict__ g) {
  size_t idx = (size_t)blockIdx.x * 256 + threadIdx.x;  // CHG*NTOK, channel-major
  int c = (int)(idx / NTOK);
  int tok = (int)(idx % NTOK);
  int y = tok / WW, x = tok % WW;
  const float* ci = convin + (size_t)c * NTOK;
  float s = cb[c];
  #pragma unroll
  for (int ky = 0; ky < 3; ++ky) {
    int ny = y + ky - 1;
    if (ny < 0 || ny >= HH) continue;
    #pragma unroll
    for (int kx = 0; kx < 3; ++kx) {
      int nx = x + kx - 1;
      if (nx < 0 || nx >= WW) continue;
      s += cw[c * 9 + ky * 3 + kx] * ci[(size_t)ny * WW + nx];
    }
  }
  float x1 = u[(size_t)tok * HID + c];
  g[(size_t)tok * CHG + c] = (_Float16)(x1 * s);
}

// ---------------- host side ----------------
extern "C" void kernel_launch(void* const* d_in, const int* in_sizes, int n_in,
                              void* d_out, int out_size, void* d_ws, size_t ws_size,
                              hipStream_t stream) {
  const float* x_tkn  = (const float*)d_in[0];
  const float* ln1_w  = (const float*)d_in[1];
  const float* ln1_b  = (const float*)d_in[2];
  const float* q1_w   = (const float*)d_in[3];
  const float* kv1_w  = (const float*)d_in[4];
  const float* rpb1   = (const float*)d_in[5];
  const float* q2_w   = (const float*)d_in[6];
  const float* kv2_w  = (const float*)d_in[7];
  const float* rpb2   = (const float*)d_in[8];
  const float* proj_w = (const float*)d_in[9];
  const float* proj_b = (const float*)d_in[10];
  const float* ln2_w  = (const float*)d_in[11];
  const float* ln2_b  = (const float*)d_in[12];
  const float* fc1_w  = (const float*)d_in[13];
  const float* fc1_b  = (const float*)d_in[14];
  const float* sgn_w  = (const float*)d_in[15];
  const float* sgn_b  = (const float*)d_in[16];
  const float* conv_w = (const float*)d_in[17];
  const float* conv_b = (const float*)d_in[18];
  const float* fc2_w  = (const float*)d_in[19];
  const float* fc2_b  = (const float*)d_in[20];
  const int*   rpi    = (const int*)d_in[21];
  float* out = (float*)d_out;

  char* ws = (char*)d_ws;
  size_t off = 0;
  auto alloc = [&](size_t bytes) -> void* {
    off = (off + 255) & ~(size_t)255;
    void* p = ws + off;
    off += bytes;
    return p;
  };

  // f16 transposed weights [N][K]
  _Float16* wq1h  = (_Float16*)alloc(64 * 64 * 2);
  _Float16* wkv1h = (_Float16*)alloc(64 * 128 * 2);
  _Float16* wq2h  = (_Float16*)alloc(64 * 64 * 2);
  _Float16* wkv2h = (_Float16*)alloc(64 * 128 * 2);
  _Float16* wprjh = (_Float16*)alloc(128 * 128 * 2);
  _Float16* wfc1h = (_Float16*)alloc(128 * 256 * 2);
  _Float16* wfc2h = (_Float16*)alloc(128 * 128 * 2);
  // activations
  _Float16* xa_h    = (_Float16*)alloc((size_t)NTOK * C2 * 2);
  _Float16* xbimg_h = (_Float16*)alloc((size_t)NTOK * C2 * 2);
  _Float16* xbp_h   = (_Float16*)alloc((size_t)NPROW * C2 * 2);
  _Float16* q1_h    = (_Float16*)alloc((size_t)NTOK * C2 * 2);
  _Float16* kv1_h   = (_Float16*)alloc((size_t)NPROW * 128 * 2);
  _Float16* q2_h    = (_Float16*)alloc((size_t)NPROW * C2 * 2);
  _Float16* kv2_h   = (_Float16*)alloc((size_t)NTOK * 128 * 2);
  _Float16* a_h     = (_Float16*)alloc((size_t)NTOK * C2 * 2);
  float*    bb_f    = (float*)alloc((size_t)NPROW * C2 * 4);
  float*    canv_f  = (float*)alloc((size_t)CANV * CANV * C2 * 4);
  _Float16* xcat_h  = (_Float16*)alloc((size_t)NTOK * DIMC * 2);
  float*    xres_f  = (float*)alloc((size_t)NTOK * DIMC * 4);
  _Float16* t_h     = (_Float16*)alloc((size_t)NTOK * DIMC * 2);
  float*    u_f     = (float*)alloc((size_t)NTOK * HID * 4);
  float*    cvin_f  = (float*)alloc((size_t)NTOK * CHG * 4);
  _Float16* g_h     = (_Float16*)alloc((size_t)NTOK * CHG * 2);

  auto cvtT = [&](const float* src, _Float16* dst, int K, int N) {
    wcvtT_k<<<(K * N + 255) / 256, 256, 0, stream>>>(src, dst, K, N);
  };
  cvtT(q1_w, wq1h, 64, 64);
  cvtT(kv1_w, wkv1h, 64, 128);
  cvtT(q2_w, wq2h, 64, 64);
  cvtT(kv2_w, wkv2h, 64, 128);
  cvtT(proj_w, wprjh, 128, 128);
  cvtT(fc1_w, wfc1h, 128, 256);
  cvtT(fc2_w, wfc2h, 128, 128);

  // LN1 + window split, then unfold overlapping patches
  ln1_part_k<<<NTOK / 4, 128, 0, stream>>>(x_tkn, ln1_w, ln1_b, xa_h, xbimg_h);
  unfold_k<<<(NPROW * 8) / 256, 256, 0, stream>>>(xbimg_h, xbp_h);

  // projections (WMMA GEMMs; BT is N x K)
  gemm_wmma_k<<<dim3(1, NTOK / 128), 256, 0, stream>>>(
      xa_h, wq1h, nullptr, nullptr, nullptr, q1_h, NTOK, 64, 64);
  gemm_wmma_k<<<dim3(2, NPROW / 128), 256, 0, stream>>>(
      xbp_h, wkv1h, nullptr, nullptr, nullptr, kv1_h, NPROW, 128, 64);
  gemm_wmma_k<<<dim3(1, NPROW / 128), 256, 0, stream>>>(
      xbp_h, wq2h, nullptr, nullptr, nullptr, q2_h, NPROW, 64, 64);
  gemm_wmma_k<<<dim3(2, NTOK / 128), 256, 0, stream>>>(
      xa_h, wkv2h, nullptr, nullptr, nullptr, kv2_h, NTOK, 128, 64);

  // cross attentions (WMMA + f16 LDS softmax + transposed V staging)
  attn_wmma_k<<<dim3(NQ1 / 16, H2, NWIN), 32, 0, stream>>>(
      q1_h, kv1_h, xa_h, rpi, rpb1, a_h, nullptr, NQ1, NK1, 1);
  attn_wmma_k<<<dim3(NK1 / 16, H2, NWIN), 32, 0, stream>>>(
      q2_h, kv2_h, xbp_h, rpi, rpb2, nullptr, bb_f, NK1, NQ1, 2);

  // fold + (bicubic resize fused into concat)
  fold_k<<<(CANV * CANV * 8) / 256, 256, 0, stream>>>(bb_f, canv_f);
  concat_k<<<(NTOK * 16) / 256, 256, 0, stream>>>(a_h, canv_f, xcat_h);

  // proj + residual
  gemm_wmma_k<<<dim3(2, NTOK / 128), 256, 0, stream>>>(
      xcat_h, wprjh, proj_b, x_tkn, xres_f, nullptr, NTOK, 128, 128);

  // LN2 -> f16
  ln_f16_k<<<NTOK / 4, 128, 0, stream>>>(xres_f, ln2_w, ln2_b, t_h);

  // fc1
  gemm_wmma_k<<<dim3(4, NTOK / 128), 256, 0, stream>>>(
      t_h, wfc1h, fc1_b, nullptr, u_f, nullptr, NTOK, 256, 128);

  // gate: LN(x2) -> NCHW, depthwise conv 3x3 + bias, multiply by x1
  sgnorm_k<<<NTOK / 4, 128, 0, stream>>>(u_f, sgn_w, sgn_b, cvin_f);
  convgate_k<<<((size_t)NTOK * CHG) / 256, 256, 0, stream>>>(cvin_f, conv_w, conv_b, u_f, g_h);

  // fc2 + residual -> final output
  gemm_wmma_k<<<dim3(2, NTOK / 128), 256, 0, stream>>>(
      g_h, wfc2h, fc2_b, xres_f, out, nullptr, NTOK, 128, 128);
}
